// Interest_GNN_6270652252670
// MI455X (gfx1250) — matrix operand order repaired and checked by reading, not compiled, not run on hardware
//
#include <hip/hip_runtime.h>
#include <hip/hip_bf16.h>

typedef __attribute__((ext_vector_type(16))) _Float16 v16h;
typedef __attribute__((ext_vector_type(8)))  _Float16 v8h;
typedef __attribute__((ext_vector_type(8)))  float    v8f;

// ---------------------------------------------------------------------------
// WMMA GEMM: out[M x NCOLS] = A[M x KDIM] (f16, row stride LDA=KPAD, zero-
// padded) * B[KDIM x NCOLS] (f32 weights) (+bias). Output f32 or f16 (OUTH),
// row stride LDO. One 16x16 tile per wave, 8 waves / 256-thread block.
// Branch-free inner loop: 2x global_load_b128 (A) + 2x ds_load_b128 (B) + wmma.
// ---------------------------------------------------------------------------
template<int KDIM, int NCOLS, int LDA, int LDO, bool BIAS, bool OUTH>
__global__ __launch_bounds__(256) void gemm_h16_wmma(
    const _Float16* __restrict__ A, const float* __restrict__ B,
    const float* __restrict__ bias, void* __restrict__ outv, int M)
{
    constexpr int KT   = (KDIM + 31) / 32;
    constexpr int KPAD = KT * 32;
    static_assert(LDA >= KPAD, "A stride must cover padded K");
    constexpr int NT   = (NCOLS + 15) / 16;
    constexpr int NPAD = NT * 16;

    // Weights in LDS, transposed [n][k], zero-padded, f16.
    __shared__ __align__(64) _Float16 bl[NPAD * KPAD];
    for (int idx = threadIdx.x; idx < NPAD * KPAD; idx += 256) {
        int n = idx / KPAD;
        int k = idx - n * KPAD;
        float v = (k < KDIM && n < NCOLS) ? B[(size_t)k * NCOLS + n] : 0.0f;
        bl[idx] = (_Float16)v;
    }
    __syncthreads();

    const int rowTiles = (M + 15) / 16;
    const int tile = blockIdx.x * 8 + (threadIdx.x >> 5);
    if (tile >= rowTiles * NT) return;            // wave-uniform
    const int rowTile = tile / NT;
    const int colTile = tile - rowTile * NT;
    const int row0 = rowTile * 16;
    const int col0 = colTile * 16;

    const int lane = threadIdx.x & 31;
    const int m    = lane & 15;
    const int hi   = lane >> 4;

    const _Float16* arow = A + (size_t)(row0 + m) * LDA;   // rows padded to 16x
    const int ncol = col0 + (lane & 15);
    const _Float16* bcol = bl + (size_t)ncol * KPAD;

    float bv = 0.0f;
    if (BIAS && ncol < NCOLS) bv = bias[ncol];
    v8f c;
    #pragma unroll
    for (int r = 0; r < 8; ++r) c[r] = bv;

    #pragma unroll
    for (int kt = 0; kt < KT; ++kt) {
        const int k0 = kt * 32;
        // A 16-bit 16x32 layout: lane-half hi -> K chunks {k0+hi*8..+7} and
        // {k0+16+hi*8..+7}; chunk element i maps linearly to fragment g*8+i.
        v8h a0 = *(const v8h*)(arow + k0 + hi * 8);
        v8h a1 = *(const v8h*)(arow + k0 + 16 + hi * 8);
        v16h a = __builtin_shufflevector(a0, a1,
                     0, 1, 2, 3, 4, 5, 6, 7, 8, 9, 10, 11, 12, 13, 14, 15);
        // B 32x16 layout: same K chunks as A ({k0+hi*8..+7}, {k0+16+hi*8..+7}),
        // from the transposed LDS tile.
        v8h b0 = *(const v8h*)(bcol + k0 + hi * 8);
        v8h b1 = *(const v8h*)(bcol + k0 + 16 + hi * 8);
        v16h b = __builtin_shufflevector(b0, b1,
                     0, 1, 2, 3, 4, 5, 6, 7, 8, 9, 10, 11, 12, 13, 14, 15);

        c = __builtin_amdgcn_wmma_f32_16x16x32_f16(
                false, a, false, b, (short)0, c, false, false);
    }

    if (ncol < NCOLS) {
        #pragma unroll
        for (int r = 0; r < 8; ++r) {
            const int row = row0 + hi * 8 + r;
            if (row < M) {
                if (OUTH) ((_Float16*)outv)[(size_t)row * LDO + ncol] = (_Float16)c[r];
                else      ((float*)outv)[(size_t)row * LDO + ncol]    = c[r];
            }
        }
    }
}

// ---------------------------------------------------------------------------
// Degree / normalization helpers
// ---------------------------------------------------------------------------
__global__ void deg_init(float* __restrict__ deg, int n) {
    int i = blockIdx.x * blockDim.x + threadIdx.x;
    if (i < n) deg[i] = 1.0f;                      // self-loop
}
__global__ void deg_count(const int* __restrict__ dst, float* __restrict__ deg, int e) {
    int i = blockIdx.x * blockDim.x + threadIdx.x;
    if (i < e) atomicAdd(&deg[dst[i]], 1.0f);
}
__global__ void deg_rsqrt(float* __restrict__ deg, int n) {
    int i = blockIdx.x * blockDim.x + threadIdx.x;
    if (i < n) deg[i] = rsqrtf(deg[i]);
}

// f32 -> zero-padded f16 (row stride dcols)
__global__ void cvt_pad_f16(const float* __restrict__ src, _Float16* __restrict__ dst,
                            int rows, int scols, int dcols) {
    int i = blockIdx.x * blockDim.x + threadIdx.x;
    if (i >= rows * dcols) return;
    int r = i / dcols;
    int c = i - r * dcols;
    dst[i] = (c < scols) ? (_Float16)src[(size_t)r * scols + c] : (_Float16)0.0f;
}

// x_h[:, 0:50] = emb_table[interest_ids]  (f16, stride 160)
__global__ void gather_emb(const int* __restrict__ ids, const float* __restrict__ table,
                           _Float16* __restrict__ xh, int n) {
    int i = blockIdx.x * blockDim.x + threadIdx.x;
    if (i >= n * 50) return;
    int node = i / 50;
    int c    = i - node * 50;
    xh[(size_t)node * 160 + c] = (_Float16)table[(size_t)ids[node] * 50 + c];
}

// agg[dst] += xw[src] * dis[src]*dis[dst]  -- one wave per edge, lanes = dims
template<int C>
__global__ void scatter_edges(const int* __restrict__ src, const int* __restrict__ dst,
                              const float* __restrict__ dis, const float* __restrict__ xw,
                              float* __restrict__ agg, int e) {
    int wave = (blockIdx.x * blockDim.x + threadIdx.x) >> 5;
    int lane = threadIdx.x & 31;
    if (wave >= e) return;
    int s = src[wave];
    int d = dst[wave];
    float norm = dis[s] * dis[d];
    const float* xr = xw + (size_t)s * C;
    float*       ar = agg + (size_t)d * C;
    for (int c = lane; c < C; c += 32)
        atomicAdd(&ar[c], xr[c] * norm);
}

// h(f16, stride LDH, pads pre-zeroed) = relu(agg + xw*dis^2 + bias)
template<int C, int LDH>
__global__ void finalize_relu(const float* __restrict__ xw, const float* __restrict__ dis,
                              const float* __restrict__ bias, const float* __restrict__ agg,
                              _Float16* __restrict__ h, int n) {
    int i = blockIdx.x * blockDim.x + threadIdx.x;
    if (i >= n * C) return;
    int node = i / C;
    int c    = i - node * C;
    float dd = dis[node];
    float v  = agg[i] + xw[i] * dd * dd + bias[c];
    h[(size_t)node * LDH + c] = (_Float16)fmaxf(v, 0.0f);
}

// ---------------------------------------------------------------------------
static inline int gemm_blocks(int M, int NCOLS) {
    int rowTiles = (M + 15) / 16;
    int NT = (NCOLS + 15) / 16;
    return (rowTiles * NT + 7) / 8;
}
static inline size_t alignup(size_t x) { return (x + 255) & ~(size_t)255; }

extern "C" void kernel_launch(void* const* d_in, const int* in_sizes, int n_in,
                              void* d_out, int out_size, void* d_ws, size_t ws_size,
                              hipStream_t stream) {
    const int*   ids   = (const int*)  d_in[0];
    const float* textF = (const float*)d_in[1];
    const int*   eidx  = (const int*)  d_in[2];
    const float* embT  = (const float*)d_in[3];
    const float* textW = (const float*)d_in[4];
    const float* textB = (const float*)d_in[5];
    const float* g1W   = (const float*)d_in[6];
    const float* g1b   = (const float*)d_in[7];
    const float* g2W   = (const float*)d_in[8];
    const float* g2b   = (const float*)d_in[9];
    const float* linW  = (const float*)d_in[10];
    const float* linb  = (const float*)d_in[11];
    float* out = (float*)d_out;

    const int N  = in_sizes[0];
    const int E  = in_sizes[2] / 2;
    const int Mp = ((N + 15) / 16) * 16;           // row-padded for tile loads
    const int* src = eidx;
    const int* dst = eidx + E;

    // Workspace layout (bytes, 256-aligned); tf16 region is dead after the
    // text GEMM and is reused for layer-2 buffers.
    char* base = (char*)d_ws;
    size_t off = 0;
    float*    dis  = (float*)   (base + off); off = alignup(off + (size_t)N  * 4);
    _Float16* xh   = (_Float16*)(base + off); off = alignup(off + (size_t)Mp * 160 * 2);
    float*    xw1  = (float*)   (base + off); off = alignup(off + (size_t)Mp * 60 * 4);
    float*    agg1 = (float*)   (base + off); off = alignup(off + (size_t)Mp * 60 * 4);
    _Float16* h1h  = (_Float16*)(base + off); off = alignup(off + (size_t)Mp * 64 * 2);
    char*     treg = base + off;               // tf16 region (Mp*320*2 bytes)
    _Float16* tf16 = (_Float16*)treg;
    float*    xw2  = (float*)treg;
    float*    agg2 = (float*)(treg + alignup((size_t)Mp * 30 * 4));
    _Float16* h2h  = (_Float16*)(treg + alignup((size_t)Mp * 30 * 4) * 2);

    const int TPB = 256;

    // 1) symmetric normalization coefficients
    deg_init <<<(N + TPB - 1) / TPB, TPB, 0, stream>>>(dis, N);
    deg_count<<<(E + TPB - 1) / TPB, TPB, 0, stream>>>(dst, dis, E);
    deg_rsqrt<<<(N + TPB - 1) / TPB, TPB, 0, stream>>>(dis, N);

    // 2) feature build: x_h = [emb_gather | text_feats @ text_W + text_b], f16
    cvt_pad_f16<<<((size_t)N * 320 + TPB - 1) / TPB, TPB, 0, stream>>>(
        textF, tf16, N, 300, 320);
    hipMemsetAsync(xh, 0, (size_t)Mp * 160 * 2, stream);   // zero K-pad cols
    gather_emb<<<((size_t)N * 50 + TPB - 1) / TPB, TPB, 0, stream>>>(ids, embT, xh, N);
    gemm_h16_wmma<300, 80, 320, 160, true, true><<<gemm_blocks(N, 80), TPB, 0, stream>>>(
        tf16, textW, textB, xh + 50, N);

    // 3) GCN layer 1
    gemm_h16_wmma<130, 60, 160, 60, false, false><<<gemm_blocks(N, 60), TPB, 0, stream>>>(
        xh, g1W, nullptr, xw1, N);
    hipMemsetAsync(agg1, 0, (size_t)Mp * 60 * 4, stream);
    scatter_edges<60><<<((size_t)E * 32 + TPB - 1) / TPB, TPB, 0, stream>>>(
        src, dst, dis, xw1, agg1, E);
    hipMemsetAsync(h1h, 0, (size_t)Mp * 64 * 2, stream);
    finalize_relu<60, 64><<<((size_t)N * 60 + TPB - 1) / TPB, TPB, 0, stream>>>(
        xw1, dis, g1b, agg1, h1h, N);

    // 4) GCN layer 2 (layer-2 buffers alias the dead tf16 region)
    gemm_h16_wmma<60, 30, 64, 30, false, false><<<gemm_blocks(N, 30), TPB, 0, stream>>>(
        h1h, g2W, nullptr, xw2, N);
    hipMemsetAsync(agg2, 0, (size_t)Mp * 30 * 4, stream);
    scatter_edges<30><<<((size_t)E * 32 + TPB - 1) / TPB, TPB, 0, stream>>>(
        src, dst, dis, xw2, agg2, E);
    hipMemsetAsync(h2h, 0, (size_t)Mp * 32 * 2, stream);
    finalize_relu<30, 32><<<((size_t)N * 30 + TPB - 1) / TPB, TPB, 0, stream>>>(
        xw2, dis, g2b, agg2, h2h, N);

    // 5) final linear: out = h2 @ lin_W + lin_b  (f32 out)
    gemm_h16_wmma<30, 18, 32, 18, true, false><<<gemm_blocks(N, 18), TPB, 0, stream>>>(
        h2h, linW, linb, out, N);
}
